// PointNet2Class_31542239822578
// MI455X (gfx1250) — compile-verified, matrix operand order checked by reading, not run on hardware
//
#include <hip/hip_runtime.h>

// ---------------------------------------------------------------------------
// PointNet++ classification for gfx1250 (MI455X). bf16 WMMA GEMMs with
// pre-transposed/padded weights so every fragment load is an unconditional
// global_load_b128 (no exec juggling in the hot loop).
// ---------------------------------------------------------------------------

typedef __attribute__((ext_vector_type(16))) __bf16 v16bf;
typedef __attribute__((ext_vector_type(8)))  __bf16 v8bf;
typedef __attribute__((ext_vector_type(8)))  float  v8f;

#define FPS_THREADS 256

// ------------------------- farthest point sampling -------------------------
__global__ void fps_kernel(const float* __restrict__ pos, int* __restrict__ cidx,
                           int N, int S) {
  __shared__ float mind[4096];
  __shared__ float rb[FPS_THREADS];
  __shared__ int   ri[FPS_THREADS];
  __shared__ float lp[3];
  __shared__ int   lastidx;
  const int b = blockIdx.x;
  const int tid = threadIdx.x;
  const float* p = pos + (size_t)b * N * 3;
  for (int i = tid; i < N; i += FPS_THREADS) mind[i] = 3.4e38f;
  if (tid == 0) lastidx = 0;
  __syncthreads();
  for (int it = 0; it < S; ++it) {
    if (tid == 0) {
      cidx[b * S + it] = lastidx;
      lp[0] = p[lastidx*3+0]; lp[1] = p[lastidx*3+1]; lp[2] = p[lastidx*3+2];
    }
    __syncthreads();
    float bv = -1.0f; int bi = 0x7fffffff;
    for (int i = tid; i < N; i += FPS_THREADS) {
      float dx = p[i*3+0]-lp[0], dy = p[i*3+1]-lp[1], dz = p[i*3+2]-lp[2];
      float d  = dx*dx + dy*dy + dz*dz;
      float md = fminf(mind[i], d);
      mind[i] = md;
      if (md > bv) { bv = md; bi = i; }  // strict > keeps first occurrence
    }
    rb[tid] = bv; ri[tid] = bi;
    __syncthreads();
    for (int s2 = FPS_THREADS/2; s2 > 0; s2 >>= 1) {
      if (tid < s2) {
        if (rb[tid+s2] > rb[tid] ||
            (rb[tid+s2] == rb[tid] && ri[tid+s2] < ri[tid])) {
          rb[tid] = rb[tid+s2]; ri[tid] = ri[tid+s2];
        }
      }
      __syncthreads();
    }
    if (tid == 0) lastidx = ri[0];
    __syncthreads();
  }
}

// ------------------------------ gather centers -----------------------------
__global__ void gather_centers(const float* __restrict__ pos,
                               const int* __restrict__ cidx,
                               float* __restrict__ centers,
                               int B, int N, int S) {
  int t = blockIdx.x * blockDim.x + threadIdx.x;
  if (t >= B * S * 3) return;
  int c = t % 3, g = t / 3;
  int b = g / S;
  centers[t] = pos[((size_t)b * N + cidx[g]) * 3 + c];
}

// ------------------------------- ball query --------------------------------
__global__ void ball_group_kernel(const float* __restrict__ pos,
                                  const float* __restrict__ centers,
                                  int* __restrict__ idx,
                                  int B, int N, int S, int K, float r2) {
  int t = blockIdx.x * blockDim.x + threadIdx.x;
  if (t >= B * S) return;
  int b = t / S;
  const float* p = pos + (size_t)b * N * 3;
  float cx = centers[t*3+0], cy = centers[t*3+1], cz = centers[t*3+2];
  int* o = idx + (size_t)t * K;
  int cnt = 0, first = 0;
  for (int i = 0; i < N && cnt < K; ++i) {
    float dx = p[i*3+0]-cx, dy = p[i*3+1]-cy, dz = p[i*3+2]-cz;
    if (dx*dx + dy*dy + dz*dz <= r2) {
      if (cnt == 0) first = i;
      o[cnt++] = i;
    }
  }
  for (int j = cnt; j < K; ++j) o[j] = first;
}

// ---------------------- weight pack: [K,N] f32 -> [Npad,Kpad] bf16 ---------
__global__ void pack_weights(const float* __restrict__ W, __bf16* __restrict__ Wt,
                             int K, int N, int Kpad, int Npad) {
  long t = (long)blockIdx.x * blockDim.x + threadIdx.x;
  if (t >= (long)Npad * Kpad) return;
  int n = (int)(t / Kpad), k = (int)(t % Kpad);
  float v = (k < K && n < N) ? W[(size_t)k * N + n] : 0.0f;
  Wt[t] = (__bf16)v;
}

// --------------------------- feature construction --------------------------
__global__ void build_feat1(const float* __restrict__ pos,
                            const float* __restrict__ cent,
                            const int* __restrict__ idx,
                            __bf16* __restrict__ F, int b0, int Bc) {
  // rows = Bc*512*32, row stride 32 (3 rel coords + zero pad)
  long t = (long)blockIdx.x * blockDim.x + threadIdx.x;
  long total = (long)Bc * 512 * 32 * 32;
  if (t >= total) return;
  int c = (int)(t & 31); long r = t >> 5;
  float v = 0.0f;
  if (c < 3) {
    int k = (int)(r % 32); int s = (int)((r / 32) % 512); int b = b0 + (int)(r / (32*512));
    int i = idx[((size_t)b * 512 + s) * 32 + k];
    v = pos[((size_t)b * 4096 + i) * 3 + c] - cent[((size_t)b * 512 + s) * 3 + c];
  }
  F[t] = (__bf16)v;
}

__global__ void build_feat2(const __bf16* __restrict__ x1,
                            const float* __restrict__ p1,
                            const float* __restrict__ cent,
                            const int* __restrict__ idx,
                            __bf16* __restrict__ F, int b0, int Bc) {
  // rows = Bc*128*64, row stride 160 (128 feat + 3 rel + zero pad)
  long t = (long)blockIdx.x * blockDim.x + threadIdx.x;
  long total = (long)Bc * 128 * 64 * 160;
  if (t >= total) return;
  int c = (int)(t % 160); long r = t / 160;
  float v = 0.0f;
  if (c < 131) {
    int k = (int)(r % 64); int s = (int)((r / 64) % 128); int b = b0 + (int)(r / (64*128));
    int i = idx[((size_t)b * 128 + s) * 64 + k];
    if (c < 128) v = (float)x1[((size_t)b * 512 + i) * 128 + c];
    else v = p1[((size_t)b * 512 + i) * 3 + (c-128)] - cent[((size_t)b * 128 + s) * 3 + (c-128)];
  }
  F[t] = (__bf16)v;
}

__global__ void build_feat3(const __bf16* __restrict__ x2,
                            const float* __restrict__ p2,
                            __bf16* __restrict__ F) {
  // rows = 32*128, row stride 288 (256 feat + 3 abs pos + zero pad)
  long t = (long)blockIdx.x * blockDim.x + threadIdx.x;
  long total = (long)32 * 128 * 288;
  if (t >= total) return;
  int c = (int)(t % 288); long r = t / 288;
  float v = 0.0f;
  if (c < 256) v = (float)x2[r * 256 + c];
  else if (c < 259) v = p2[r * 3 + (c-256)];
  F[t] = (__bf16)v;
}

// ------------------------ WMMA GEMM + fused epilogue -----------------------
// D = act(A[M,Kpad](bf16, zero-padded) x Wt[Npad,Kpad](bf16, transposed)).
// M multiple of 16, Kpad multiple of 32, Npad multiple of 16 -> no bounds
// checks anywhere in the hot loop; all fragment loads are b128.
// flags: 1=relu, 4=write f32, 8=bn-before-relu (head ordering).
__global__ void gemm_wmma(const __bf16* __restrict__ A, int Kpad,
                          const __bf16* __restrict__ Wt,
                          const float* __restrict__ bias,
                          const float* __restrict__ gamma,
                          const float* __restrict__ beta,
                          int N, int Npad,
                          void* __restrict__ out, int flags) {
  const int lane  = threadIdx.x;            // 0..31
  const int mn    = lane & 15;
  const int grp   = lane >> 4;
  const int kb0   = grp << 3;               // 0 or 8
  const int mbase = blockIdx.x * 16;
  const int nbase = blockIdx.y * 16;
  const __bf16* Ap = A  + (size_t)(mbase + mn) * Kpad;   // lane's A row
  const __bf16* Bp = Wt + (size_t)(nbase + mn) * Kpad;   // lane's B column (row of Wt)

  v8f acc = {};
  const int ktiles = Kpad >> 5;
  for (int kt = 0; kt < ktiles; ++kt) {
    const int k = (kt << 5) + kb0;
    // 16-bit fragment layout: lane holds k in {kb0..kb0+7} U {kb0+16..kb0+23}
    v8bf alo = *(const v8bf*)(Ap + k);
    v8bf ahi = *(const v8bf*)(Ap + k + 16);
    v8bf blo = *(const v8bf*)(Bp + k);
    v8bf bhi = *(const v8bf*)(Bp + k + 16);
    v16bf a = __builtin_shufflevector(alo, ahi, 0,1,2,3,4,5,6,7,8,9,10,11,12,13,14,15);
    v16bf b = __builtin_shufflevector(blo, bhi, 0,1,2,3,4,5,6,7,8,9,10,11,12,13,14,15);
    acc = __builtin_amdgcn_wmma_f32_16x16x32_bf16(false, a, false, b,
                                                  (short)0, acc, false, false);
  }

  const int col = nbase + mn;
  const int cc  = (col < N) ? col : 0;      // clamp (pad cols get garbage, never read)
  float bi = bias ? bias[cc] : 0.0f;
  float sc = 1.0f, be = 0.0f;
  if (gamma) { sc = gamma[cc] * rsqrtf(1.0f + 1e-5f); be = beta[cc]; }
#pragma unroll
  for (int v = 0; v < 8; ++v) {
    int r = mbase + v + 8 * grp;            // C/D layout: lane = n + 16*(m>>3), vgpr = m&7
    float z = acc[v] + bi;
    if (flags & 8) {                        // head: bn then relu
      if (gamma) z = z * sc + be;
      if (flags & 1) z = fmaxf(z, 0.0f);
    } else {                                // mlp_apply: relu then bn
      if (flags & 1) z = fmaxf(z, 0.0f);
      if (gamma) z = z * sc + be;
    }
    if (flags & 4) ((float*)out)[(size_t)r * Npad + col] = z;
    else ((__bf16*)out)[(size_t)r * Npad + col] = (__bf16)z;
  }
}

// -------------------------------- max pooling ------------------------------
__global__ void maxpool_kernel(const __bf16* __restrict__ in,
                               __bf16* __restrict__ out,
                               int groups, int K, int C) {
  long t = (long)blockIdx.x * blockDim.x + threadIdx.x;
  if (t >= (long)groups * C) return;
  long g = t / C; int c = (int)(t % C);
  float m = -3.4e38f;
  const __bf16* base = in + ((size_t)g * K) * C + c;
  for (int k = 0; k < K; ++k) m = fmaxf(m, (float)base[(size_t)k * C]);
  out[t] = (__bf16)m;
}

// -------------------------------- log softmax ------------------------------
__global__ void logsoftmax_kernel(const float* __restrict__ logits, int stride,
                                  float* __restrict__ out, int C) {
  __shared__ float buf[64];
  __shared__ float mx, lse;
  int b = blockIdx.x, tid = threadIdx.x;
  if (tid < C) buf[tid] = logits[b * stride + tid];
  __syncthreads();
  if (tid == 0) {
    float m = buf[0];
    for (int i = 1; i < C; ++i) m = fmaxf(m, buf[i]);
    float s = 0.0f;
    for (int i = 0; i < C; ++i) s += expf(buf[i] - m);
    mx = m; lse = logf(s);
  }
  __syncthreads();
  if (tid < C) out[b * C + tid] = buf[tid] - mx - lse;
}

// ---------------------------------------------------------------------------
extern "C" void kernel_launch(void* const* d_in, const int* in_sizes, int n_in,
                              void* d_out, int out_size, void* d_ws, size_t ws_size,
                              hipStream_t stream) {
  (void)in_sizes; (void)n_in; (void)out_size; (void)ws_size;
  const float* pos = (const float*)d_in[0];
  auto P = [&](int base, int l, int f) { return (const float*)d_in[base + 4*l + f]; };
  const int M1 = 1, M2 = 13, M3 = 25;
  const float* hW1 = (const float*)d_in[37]; const float* hb1 = (const float*)d_in[38];
  const float* hg1 = (const float*)d_in[39]; const float* he1 = (const float*)d_in[40];
  const float* hW2 = (const float*)d_in[41]; const float* hb2 = (const float*)d_in[42];
  const float* hg2 = (const float*)d_in[43]; const float* he2 = (const float*)d_in[44];
  const float* hW3 = (const float*)d_in[45]; const float* hb3 = (const float*)d_in[46];

  // ---- workspace bump allocator ----
  char* wp = (char*)d_ws;
  auto alloc = [&](size_t bytes) -> void* {
    void* p = (void*)wp; wp += (bytes + 255) & ~(size_t)255; return p;
  };
  int*    cidx1 = (int*)   alloc((size_t)32*512*4);
  float*  cent1 = (float*) alloc((size_t)32*512*3*4);
  int*    idx1  = (int*)   alloc((size_t)32*512*32*4);
  int*    cidx2 = (int*)   alloc((size_t)32*128*4);
  float*  cent2 = (float*) alloc((size_t)32*128*3*4);
  int*    idx2  = (int*)   alloc((size_t)32*128*64*4);
  __bf16* x1    = (__bf16*)alloc((size_t)32*512*128*2);
  __bf16* x2    = (__bf16*)alloc((size_t)32*128*256*2);
  __bf16* gbuf  = (__bf16*)alloc((size_t)32*1024*2);
  __bf16* h1    = (__bf16*)alloc((size_t)32*512*2);
  __bf16* h2    = (__bf16*)alloc((size_t)32*256*2);
  float*  logit = (float*) alloc((size_t)32*48*4);
  __bf16* F     = (__bf16*)alloc((size_t)5242880*2);  // max feat chunk (feat2: 32768*160)
  __bf16* P0    = (__bf16*)alloc((size_t)8388608*2);  // max interm chunk (65536*128)
  __bf16* P1    = (__bf16*)alloc((size_t)8388608*2);

  // ---- pack all weights: transposed + padded bf16 ----
  struct WDesc { const float* W; int K, N, Kpad, Npad; __bf16* Wt; };
  WDesc wd[12] = {
    { P(M1,0,0),    3,   64,   32,   64, nullptr },
    { P(M1,1,0),   64,   64,   64,   64, nullptr },
    { P(M1,2,0),   64,  128,   64,  128, nullptr },
    { P(M2,0,0),  131,  128,  160,  128, nullptr },
    { P(M2,1,0),  128,  128,  128,  128, nullptr },
    { P(M2,2,0),  128,  256,  128,  256, nullptr },
    { P(M3,0,0),  259,  256,  288,  256, nullptr },
    { P(M3,1,0),  256,  512,  256,  512, nullptr },
    { P(M3,2,0),  512, 1024,  512, 1024, nullptr },
    { hW1,       1024,  512, 1024,  512, nullptr },
    { hW2,        512,  256,  512,  256, nullptr },
    { hW3,        256,   40,  256,   48, nullptr },
  };
  auto cdiv = [](long a, long b) { return (unsigned)((a + b - 1) / b); };
  for (int i = 0; i < 12; ++i) {
    long elems = (long)wd[i].Npad * wd[i].Kpad;
    wd[i].Wt = (__bf16*)alloc((size_t)elems * 2);
    pack_weights<<<cdiv(elems, 256), 256, 0, stream>>>(
        wd[i].W, wd[i].Wt, wd[i].K, wd[i].N, wd[i].Kpad, wd[i].Npad);
  }

  auto gemm = [&](const __bf16* A, int Kpad, int M, int wi,
                  const float* bias, const float* g, const float* be,
                  void* out, int flags) {
    dim3 grid(M / 16, wd[wi].Npad / 16);
    gemm_wmma<<<grid, 32, 0, stream>>>(A, Kpad, wd[wi].Wt, bias, g, be,
                                       wd[wi].N, wd[wi].Npad, out, flags);
  };

  const int CB = 4;  // batch chunk

  // ---------------- SA1: S=512, K=32, r=0.2 ----------------
  fps_kernel<<<32, FPS_THREADS, 0, stream>>>(pos, cidx1, 4096, 512);
  gather_centers<<<cdiv((long)32*512*3, 256), 256, 0, stream>>>(pos, cidx1, cent1, 32, 4096, 512);
  ball_group_kernel<<<cdiv((long)32*512, 256), 256, 0, stream>>>(pos, cent1, idx1, 32, 4096, 512, 32, 0.04f);
  for (int b0 = 0; b0 < 32; b0 += CB) {
    const int rows = CB * 512 * 32;  // 65536
    build_feat1<<<cdiv((long)rows*32, 256), 256, 0, stream>>>(pos, cent1, idx1, F, b0, CB);
    gemm(F,   32, rows, 0, P(M1,0,1), P(M1,0,2), P(M1,0,3), P0, 1);
    gemm(P0,  64, rows, 1, P(M1,1,1), P(M1,1,2), P(M1,1,3), P1, 1);
    gemm(P1,  64, rows, 2, P(M1,2,1), P(M1,2,2), P(M1,2,3), P0, 1);
    maxpool_kernel<<<cdiv((long)CB*512*128, 256), 256, 0, stream>>>(
        P0, x1 + (size_t)b0*512*128, CB*512, 32, 128);
  }

  // ---------------- SA2: S=128, K=64, r=0.4 ----------------
  fps_kernel<<<32, FPS_THREADS, 0, stream>>>(cent1, cidx2, 512, 128);
  gather_centers<<<cdiv((long)32*128*3, 256), 256, 0, stream>>>(cent1, cidx2, cent2, 32, 512, 128);
  ball_group_kernel<<<cdiv((long)32*128, 256), 256, 0, stream>>>(cent1, cent2, idx2, 32, 512, 128, 64, 0.16f);
  for (int b0 = 0; b0 < 32; b0 += CB) {
    const int rows = CB * 128 * 64;  // 32768
    build_feat2<<<cdiv((long)rows*160, 256), 256, 0, stream>>>(x1, cent1, cent2, idx2, F, b0, CB);
    gemm(F,  160, rows, 3, P(M2,0,1), P(M2,0,2), P(M2,0,3), P0, 1);
    gemm(P0, 128, rows, 4, P(M2,1,1), P(M2,1,2), P(M2,1,3), P1, 1);
    gemm(P1, 128, rows, 5, P(M2,2,1), P(M2,2,2), P(M2,2,3), P0, 1);
    maxpool_kernel<<<cdiv((long)CB*128*256, 256), 256, 0, stream>>>(
        P0, x2 + (size_t)b0*128*256, CB*128, 64, 256);
  }

  // ---------------- Global SA: MLP[259,256,512,1024] + max over S=128 ------
  build_feat3<<<cdiv((long)32*128*288, 256), 256, 0, stream>>>(x2, cent2, F);
  {
    const int rows = 32 * 128;  // 4096
    gemm(F,  288, rows, 6, P(M3,0,1), P(M3,0,2), P(M3,0,3), P0, 1);
    gemm(P0, 256, rows, 7, P(M3,1,1), P(M3,1,2), P(M3,1,3), P1, 1);
    gemm(P1, 512, rows, 8, P(M3,2,1), P(M3,2,2), P(M3,2,3), P0, 1);
    maxpool_kernel<<<cdiv((long)32*1024, 256), 256, 0, stream>>>(P0, gbuf, 32, 128, 1024);
  }

  // ---------------- Head: relu(bn(x@W+b)) x2, then linear + log_softmax ----
  gemm(gbuf, 1024, 32,  9, hb1, hg1, he1, h1, 1 | 8);
  gemm(h1,    512, 32, 10, hb2, hg2, he2, h2, 1 | 8);
  gemm(h2,    256, 32, 11, hb3, nullptr, nullptr, logit, 4);
  logsoftmax_kernel<<<32, 64, 0, stream>>>(logit, 48, (float*)d_out, 40);
}
